// UpstreamExpert_88287347736631
// MI455X (gfx1250) — matrix operand order repaired and checked by reading, not compiled, not run on hardware
//
#include <hip/hip_runtime.h>
#include <hip/hip_bf16.h>

// ---------------------------------------------------------------------------
// VQ + embedding gather for MI455X (gfx1250, wave32, WMMA).
//   X  = hs  [32768 x 768] f32
//   C  =     [768 x 1000]  f32 (codebook, column-per-cluster)
//   ids[n] = argmin_k ( Cnorm[k] - 2 * (X @ C)[n,k] )   (||x||^2 dropped)
//   out[n*256+e] = emb_table[ids[n]*256 + e]            (stack reshape = view)
//
// Round 2: M=32 per wave (two register-resident A tiles) so every B fragment
// loaded from L2 feeds TWO WMMAs -> halves L2 traffic and VMEM-per-WMMA.
// ---------------------------------------------------------------------------

typedef _Float16 v16h __attribute__((ext_vector_type(16)));
typedef float    v8f  __attribute__((ext_vector_type(8)));

#define D_DIM   768
#define K_DIM   1000
#define K_PAD   1008          // 63 tiles of 16
#define N_TILES 63
#define K_CHUNK 24            // 768 / 32
#define N_ROWS  32768

// ---------------------------------------------------------------------------
// Kernel 1: build Ct_h [K_PAD][768] f16 (transposed, f16) and Cnorm [K_PAD].
// grid = 63 blocks x 256 threads; block handles 16 columns of C.
// ---------------------------------------------------------------------------
__global__ __launch_bounds__(256)
void vq_prep_kernel(const float* __restrict__ C,
                    _Float16* __restrict__ Cth,
                    float* __restrict__ Cnorm)
{
    const int t  = threadIdx.x;
    const int tx = t & 15;          // column within tile
    const int ty = t >> 4;          // d-parallel group
    const int k  = blockIdx.x * 16 + tx;

    float sum = 0.0f;
    for (int d = ty; d < D_DIM; d += 16) {
        float v = (k < K_DIM) ? C[d * K_DIM + k] : 0.0f;
        Cth[(size_t)k * D_DIM + d] = (_Float16)v;
        sum += v * v;
    }

    __shared__ float red[256];
    red[t] = sum;
    __syncthreads();
    if (ty == 0) {
        float s = 0.0f;
        #pragma unroll
        for (int j = 0; j < 16; ++j) s += red[j * 16 + tx];
        Cnorm[k] = (k < K_DIM) ? s : 3.0e38f;
    }
}

// ---------------------------------------------------------------------------
// Kernel 2: WMMA GEMM (f16 in, f32 accum) fused with online argmin.
// 128 threads = 4 waves per block; one wave owns TWO 16-row M-tiles (M=32).
// Both A tiles (2 x 16x768) converted f32->f16 once into 48 register-resident
// v16h fragments (~384 VGPRs); each streamed B fragment feeds two WMMAs.
// ---------------------------------------------------------------------------
__global__ __launch_bounds__(128)
void vq_argmin_kernel(const float* __restrict__ X,
                      const _Float16* __restrict__ Cth,
                      const float* __restrict__ Cnorm,
                      int* __restrict__ ids)
{
    const int lane  = threadIdx.x & 31;
    const int wave  = blockIdx.x * 4 + (threadIdx.x >> 5);
    const int mBase = wave * 32;
    const int half  = lane >> 4;    // 0: lanes 0-15, 1: lanes 16-31
    const int l15   = lane & 15;

    // ---- Load + convert both A tiles into WMMA fragment layout (registers).
    // 16-bit A 16x32: lane row = lane&15; halves 0..7 <- k = c*32 + half*8 + j,
    //                 halves 8..15 <- +16.
    v16h a[2][K_CHUNK];
    #pragma unroll
    for (int m = 0; m < 2; ++m) {
        const float* xrow =
            X + (size_t)(mBase + m * 16 + l15) * D_DIM + half * 8;
        #pragma unroll
        for (int c = 0; c < K_CHUNK; ++c) {
            const float* p = xrow + c * 32;
            float4 f0 = *(const float4*)(p);
            float4 f1 = *(const float4*)(p + 4);
            float4 f2 = *(const float4*)(p + 16);
            float4 f3 = *(const float4*)(p + 20);
            v16h v;
            v[0]  = (_Float16)f0.x; v[1]  = (_Float16)f0.y;
            v[2]  = (_Float16)f0.z; v[3]  = (_Float16)f0.w;
            v[4]  = (_Float16)f1.x; v[5]  = (_Float16)f1.y;
            v[6]  = (_Float16)f1.z; v[7]  = (_Float16)f1.w;
            v[8]  = (_Float16)f2.x; v[9]  = (_Float16)f2.y;
            v[10] = (_Float16)f2.z; v[11] = (_Float16)f2.w;
            v[12] = (_Float16)f3.x; v[13] = (_Float16)f3.y;
            v[14] = (_Float16)f3.z; v[15] = (_Float16)f3.w;
            a[m][c] = v;
        }
    }

    float bestVal[2][8];
    int   bestIdx[2][8];
    #pragma unroll
    for (int m = 0; m < 2; ++m)
        #pragma unroll
        for (int r = 0; r < 8; ++r) { bestVal[m][r] = 3.4e38f; bestIdx[m][r] = 0; }

    // ---- N-tile loop over 63 column tiles of the codebook.
    for (int nt = 0; nt < N_TILES; ++nt) {
        v8f acc0 = {};
        v8f acc1 = {};
        // 16-bit B 32x16: lane col = lane&15; halves 0..7 <- k = c*32 + half*8 + j,
        // halves 8..15 <- +16 (matches A).
        const _Float16* brow =
            Cth + (size_t)(nt * 16 + l15) * D_DIM + half * 8;
        #pragma unroll
        for (int c = 0; c < K_CHUNK; ++c) {
            typedef _Float16 v8h_t __attribute__((ext_vector_type(8)));
            v8h_t b0 = *(const v8h_t*)(brow + c * 32);
            v8h_t b1 = *(const v8h_t*)(brow + c * 32 + 16);
            v16h b = __builtin_shufflevector(b0, b1, 0, 1, 2, 3, 4, 5, 6, 7, 8, 9, 10, 11, 12, 13, 14, 15);
            acc0 = __builtin_amdgcn_wmma_f32_16x16x32_f16(
                false, a[0][c], false, b, (short)0, acc0, false, false);
            acc1 = __builtin_amdgcn_wmma_f32_16x16x32_f16(
                false, a[1][c], false, b, (short)0, acc1, false, false);
        }

        const int   col = nt * 16 + l15;
        const float cn  = Cnorm[col];
        // accM[r] holds S[M = mBase + m*16 + r + 8*half, N = col].
        #pragma unroll
        for (int r = 0; r < 8; ++r) {
            float s0 = fmaf(-2.0f, acc0[r], cn);
            if (s0 < bestVal[0][r]) { bestVal[0][r] = s0; bestIdx[0][r] = col; }
            float s1 = fmaf(-2.0f, acc1[r], cn);
            if (s1 < bestVal[1][r]) { bestVal[1][r] = s1; bestIdx[1][r] = col; }
        }
    }

    // ---- Reduce over the 16 lanes holding each row (first-min tie-break).
    #pragma unroll
    for (int m = 0; m < 2; ++m) {
        #pragma unroll
        for (int r = 0; r < 8; ++r) {
            float v  = bestVal[m][r];
            int   ix = bestIdx[m][r];
            #pragma unroll
            for (int off = 1; off < 16; off <<= 1) {
                float ov = __shfl_xor(v, off, 32);
                int   oi = __shfl_xor(ix, off, 32);
                if (ov < v || (ov == v && oi < ix)) { v = ov; ix = oi; }
            }
            if (l15 == 0) ids[mBase + m * 16 + half * 8 + r] = ix;
        }
    }
}

// ---------------------------------------------------------------------------
// Kernel 3: coalesced float4 gather  out[n][:] = emb_table[ids[n]][:]
// grid = 8192 x 256; each thread moves one float4 (256 floats/row = 64 f4).
// ---------------------------------------------------------------------------
__global__ __launch_bounds__(256)
void vq_gather_kernel(const int* __restrict__ ids,
                      const float* __restrict__ emb,
                      float* __restrict__ out)
{
    const int t   = blockIdx.x * 256 + threadIdx.x;   // float4 index
    const int row = t >> 6;
    const int e4  = t & 63;
    const int id  = ids[row];
    ((float4*)out)[t] = ((const float4*)emb)[(size_t)id * 64 + e4];
}

// ---------------------------------------------------------------------------
extern "C" void kernel_launch(void* const* d_in, const int* in_sizes, int n_in,
                              void* d_out, int out_size, void* d_ws, size_t ws_size,
                              hipStream_t stream)
{
    (void)in_sizes; (void)n_in; (void)out_size; (void)ws_size;

    const float* hs  = (const float*)d_in[0];   // [8,4096,768]
    const float* C   = (const float*)d_in[1];   // [768,1000]
    const float* emb = (const float*)d_in[2];   // [1000,256]
    // d_in[3] = stack_num (=2) is absorbed into the output layout (pure view).

    char* ws = (char*)d_ws;
    _Float16* Cth   = (_Float16*)ws;                                  // 1.55 MB
    float*    Cnorm = (float*)(ws + (size_t)K_PAD * D_DIM * 2);       // 4 KB
    int*      ids   = (int*)(ws + (size_t)K_PAD * D_DIM * 2 + K_PAD * 4);

    vq_prep_kernel  <<<N_TILES, 256, 0, stream>>>(C, Cth, Cnorm);
    vq_argmin_kernel<<<N_ROWS / 128, 128, 0, stream>>>(hs, Cth, Cnorm, ids);
    vq_gather_kernel<<<(N_ROWS * 64) / 256, 256, 0, stream>>>(ids, emb, (float*)d_out);
}
